// ONINorm_12421045420976
// MI455X (gfx1250) — compile-verified, matrix operand order checked by reading, not run on hardware
//
#include <hip/hip_runtime.h>

// ---------------- types ----------------
typedef __attribute__((ext_vector_type(16))) __bf16 v16bf;
typedef __attribute__((ext_vector_type(8)))  float  v8f;
typedef __attribute__((ext_vector_type(4)))  float  vf4;

union Frag  { v16bf v; vf4 f[2]; };
union Chunk { vf4 f; __bf16 h[8]; };

#define N_DIM 2048
#define K_FAN 18432           // 2048*3*3
#define LDP 40                // [col][k] pitch in bf16 (16B aligned rows; 20-bank stride)

// ---------------- CDNA5 async copy: global -> LDS, 16B per lane ----------------
__device__ __forceinline__ void async_copy_b128(unsigned lds_off, const void* gptr) {
    // GLOBAL_LOAD_ASYNC_TO_LDS_B128 (GV mode): copies 16B from global address vaddr to LDS offset vdst
    asm volatile("global_load_async_to_lds_b128 %0, %1, off"
                 :: "v"(lds_off), "v"(gptr) : "memory");
}
__device__ __forceinline__ void wait_async0() {
    asm volatile("s_wait_asynccnt 0x0" ::: "memory");
}
__device__ __forceinline__ unsigned lds_addr32(const void* p) {
    // flat LDS address: offset lives in addr[31:0]
    return (unsigned)(unsigned long long)p;
}

// ---------------- reduction: sum of squares (deterministic two-pass) ----------------
__global__ __launch_bounds__(256) void sumsq_partial(const float* __restrict__ z,
                                                     size_t n, float* __restrict__ partials) {
    size_t i = (size_t)blockIdx.x * blockDim.x + threadIdx.x;
    size_t stride = (size_t)gridDim.x * blockDim.x;
    float s = 0.f;
    for (; i < n; i += stride) { float v = z[i]; s += v * v; }
    for (int off = 16; off > 0; off >>= 1) s += __shfl_down(s, off, 32);
    __shared__ float red[8];
    int lane = threadIdx.x & 31, wid = threadIdx.x >> 5;
    if (lane == 0) red[wid] = s;
    __syncthreads();
    if (threadIdx.x == 0) {
        float t = 0.f;
        for (int w = 0; w < 8; ++w) t += red[w];
        partials[blockIdx.x] = t;
    }
}

__global__ __launch_bounds__(256) void sumsq_final(const float* __restrict__ partials,
                                                   int n, float* __restrict__ out) {
    float s = 0.f;
    for (int i = threadIdx.x; i < n; i += 256) s += partials[i];
    for (int off = 16; off > 0; off >>= 1) s += __shfl_down(s, off, 32);
    __shared__ float red[8];
    int lane = threadIdx.x & 31, wid = threadIdx.x >> 5;
    if (lane == 0) red[wid] = s;
    __syncthreads();
    if (threadIdx.x == 0) {
        float t = 0.f;
        for (int w = 0; w < 8; ++w) t += red[w];
        *out = t;
    }
}

// ---------------- V = Z * rsqrt(||Z||^2), stored bf16 ----------------
__global__ __launch_bounds__(256) void scale_to_bf16(const float* __restrict__ z, size_t n,
                                                     const float* __restrict__ normsq,
                                                     unsigned short* __restrict__ vq) {
    __bf16* v = (__bf16*)vq;
    float rs = rsqrtf(*normsq);
    size_t i = (size_t)blockIdx.x * blockDim.x + threadIdx.x;
    size_t stride = (size_t)gridDim.x * blockDim.x;
    for (; i < n; i += stride) v[i] = (__bf16)(z[i] * rs);
}

// ---------------- B = I (bf16) ----------------
__global__ __launch_bounds__(256) void init_eye_bf16(unsigned short* __restrict__ bq, int n) {
    __bf16* b = (__bf16*)bq;
    size_t i = (size_t)blockIdx.x * blockDim.x + threadIdx.x;
    size_t total = (size_t)n * n;
    size_t stride = (size_t)gridDim.x * blockDim.x;
    for (; i < total; i += stride) {
        int r = (int)(i / n), c = (int)(i % n);
        b[i] = (__bf16)((r == c) ? 1.f : 0.f);
    }
}

// ---------------- bf16 WMMA GEMM, fully async-LDS, double-buffered ----------------
// C[M,N] = alpha * A @ op(B) + beta * D + epsdiag*I
//   TRANSB=false : B stored [K,N] row-major  (async to natural tile + LDS transpose)
//   TRANSB=true  : B stored [N,K] row-major  (async direct to [col][k] tile)
// Block tile 128x128, 256 threads (8 waves), wave tile 32x64 (2x4 WMMA accs), K-step 32.
// M,N multiples of 128; K multiple of 32.
template <bool TRANSB>
__global__ __launch_bounds__(256) void gemm_bf16(
        const unsigned short* __restrict__ Aq, const unsigned short* __restrict__ Bq,
        const unsigned short* __restrict__ Dq,      // bf16, may be null
        unsigned short* __restrict__ outBfq,        // bf16, may be null
        float* __restrict__ outF,                   // f32,  may be null
        int M, int N, int K, float alpha, float beta, float epsdiag) {
    const __bf16* A = (const __bf16*)Aq;
    const __bf16* B = (const __bf16*)Bq;
    const __bf16* D = (const __bf16*)Dq;
    __bf16* outBf = (__bf16*)outBfq;

    __shared__ __bf16 As[2][128 * LDP];                    // [buf][row][k]
    __shared__ __bf16 Bs[2][128 * LDP];                    // [buf][col][k]
    __shared__ __bf16 Nat[TRANSB ? 2 : 2 * 32 * 128];      // [buf][k][n] natural B tile (nn only)

    const int tid  = threadIdx.x;
    const int lane = tid & 31;
    const int wid  = tid >> 5;
    const int wm   = wid & 3;          // 0..3 : 32-row slab
    const int wn   = wid >> 2;         // 0..1 : 64-col slab
    const int bm   = blockIdx.y * 128;
    const int bn   = blockIdx.x * 128;
    const int lr   = lane & 15;        // row/col within 16
    const int hsel = (lane >> 4) << 3; // 0 or 8: K-subgroup per documented bf16 layout

    // ---- staging: all global traffic via async engine ----
    auto stageA = [&](int buf, int kk) {
        #pragma unroll
        for (int u = 0; u < 2; ++u) {
            int c = tid + u * 256;                 // 512 chunks of 8 halves
            int r = c >> 2, cs = (c & 3) << 3;
            async_copy_b128(lds_addr32(&As[buf][r * LDP + cs]),
                            A + (size_t)(bm + r) * K + kk + cs);
        }
    };
    auto stageB = [&](int buf, int kk) {
        if (TRANSB) {
            #pragma unroll
            for (int u = 0; u < 2; ++u) {
                int c = tid + u * 256;
                int n = c >> 2, ks = (c & 3) << 3;
                async_copy_b128(lds_addr32(&Bs[buf][n * LDP + ks]),
                                B + (size_t)(bn + n) * K + kk + ks);
            }
        } else {
            // natural [k][n] layout, perfectly coalesced
            #pragma unroll
            for (int u = 0; u < 2; ++u) {
                int c = tid + u * 256;
                int k = c >> 4, ns = (c & 15) << 3;
                async_copy_b128(lds_addr32(&Nat[(buf * 32 + k) * 128 + ns]),
                                B + (size_t)(kk + k) * N + bn + ns);
            }
        }
    };
    // LDS->LDS transpose: Nat[buf][k][n] -> Bs[buf][n][k] (conflict-free both sides)
    auto transB = [&](int buf) {
        int n = tid & 127, kh = tid >> 7;          // kh = 0/1 -> k range [16kh, 16kh+16)
        Chunk lo, hi;
        #pragma unroll
        for (int q = 0; q < 8; ++q) {
            lo.h[q] = Nat[(buf * 32 + kh * 16 + q) * 128 + n];
            hi.h[q] = Nat[(buf * 32 + kh * 16 + 8 + q) * 128 + n];
        }
        *(vf4*)(&Bs[buf][n * LDP + kh * 16])     = lo.f;
        *(vf4*)(&Bs[buf][n * LDP + kh * 16 + 8]) = hi.f;
    };

    v8f acc[2][4];
    #pragma unroll
    for (int i = 0; i < 2; ++i)
        #pragma unroll
        for (int j = 0; j < 4; ++j)
            #pragma unroll
            for (int r = 0; r < 8; ++r) acc[i][j][r] = 0.f;

    const int nsteps = K >> 5;
    stageA(0, 0);
    stageB(0, 0);

    for (int s = 0; s < nsteps; ++s) {
        const int buf = s & 1;
        wait_async0();          // this wave's async copies landed
        __syncthreads();        // everyone's copies visible

        if (s + 1 < nsteps) {   // overlap next tile's copies with this tile's work
            stageA(buf ^ 1, (s + 1) << 5);
            stageB(buf ^ 1, (s + 1) << 5);
        }
        if (!TRANSB) {
            transB(buf);        // build [col][k] tile for THIS step
            __syncthreads();    // transposed tile visible to all waves
        }

        // ---- fragments: two contiguous 16B LDS loads each (documented bf16 layout) ----
        Frag af[2], bfr[4];
        #pragma unroll
        for (int i = 0; i < 2; ++i) {
            int row = wm * 32 + i * 16 + lr;
            af[i].f[0] = *(const vf4*)(&As[buf][row * LDP + hsel]);      // K = hsel..hsel+7
            af[i].f[1] = *(const vf4*)(&As[buf][row * LDP + 16 + hsel]); // K = 16+hsel..
        }
        #pragma unroll
        for (int j = 0; j < 4; ++j) {
            int col = wn * 64 + j * 16 + lr;
            bfr[j].f[0] = *(const vf4*)(&Bs[buf][col * LDP + hsel]);
            bfr[j].f[1] = *(const vf4*)(&Bs[buf][col * LDP + 16 + hsel]);
        }
        #pragma unroll
        for (int i = 0; i < 2; ++i)
            #pragma unroll
            for (int j = 0; j < 4; ++j)
                acc[i][j] = __builtin_amdgcn_wmma_f32_16x16x32_bf16(
                    false, af[i].v, false, bfr[j].v, (short)0, acc[i][j], false, false);
        __syncthreads();        // all reads of buf done before it is overwritten
    }

    // ---- epilogue: C/D layout -> VGPR r holds M = r + 8*(lane>=16), N = lane%16 ----
    const int rbase = (lane >> 4) << 3;
    #pragma unroll
    for (int i = 0; i < 2; ++i) {
        #pragma unroll
        for (int j = 0; j < 4; ++j) {
            #pragma unroll
            for (int r = 0; r < 8; ++r) {
                int m = bm + wm * 32 + i * 16 + rbase + r;
                int n = bn + wn * 64 + j * 16 + lr;
                size_t idx = (size_t)m * N + n;
                float v = alpha * acc[i][j][r];
                if (D) v += beta * (float)D[idx];
                if (epsdiag != 0.f && m == n) v += epsdiag;
                if (outBf) outBf[idx] = (__bf16)v;
                if (outF)  outF[idx]  = v;
            }
        }
    }
}

// ---------------- host orchestration ----------------
extern "C" void kernel_launch(void* const* d_in, const int* in_sizes, int n_in,
                              void* d_out, int out_size, void* d_ws, size_t ws_size,
                              hipStream_t stream) {
    (void)in_sizes; (void)n_in; (void)out_size; (void)ws_size;
    const float* Z = (const float*)d_in[0];
    float* Wout = (float*)d_out;

    const int    Nn = N_DIM;
    const int    Kf = K_FAN;
    const size_t nZ = (size_t)Nn * Kf;        // 37,748,736
    const size_t nS = (size_t)Nn * Nn;        //  4,194,304

    char* ws = (char*)d_ws;
    float* partials = (float*)ws;                       // 2048 floats (8 KB)
    float* normsq   = (float*)(ws + 8192);              // 1 float
    char*  p        = ws + 16384;
    unsigned short* Vb = (unsigned short*)p; p += nZ * 2;
    unsigned short* Sb = (unsigned short*)p; p += nS * 2;
    unsigned short* B0 = (unsigned short*)p; p += nS * 2;
    unsigned short* B1 = (unsigned short*)p; p += nS * 2;
    unsigned short* T1 = (unsigned short*)p; p += nS * 2;
    unsigned short* T2 = (unsigned short*)p; p += nS * 2;

    // 1) ||Z||^2 (deterministic two-pass)
    sumsq_partial<<<2048, 256, 0, stream>>>(Z, nZ, partials);
    sumsq_final  <<<1,    256, 0, stream>>>(partials, 2048, normsq);

    // 2) V = Z * rsqrt(||Z||^2)  (bf16)
    scale_to_bf16<<<2048, 256, 0, stream>>>(Z, nZ, normsq, Vb);

    // 3) B = I
    init_eye_bf16<<<4096, 256, 0, stream>>>(B0, Nn);

    // 4) S = V V^T + eps I   (A=V [N,K], B=V [N,K] transposed)
    dim3 gS(Nn / 128, Nn / 128);
    gemm_bf16<true><<<gS, 256, 0, stream>>>(Vb, Vb, nullptr, Sb, nullptr,
                                            Nn, Nn, Kf, 1.0f, 0.0f, 1e-5f);

    // 5) 5 Newton-Schulz steps: B' = 1.5 B - 0.5 (B*B*B) S
    unsigned short* cur = B0;
    unsigned short* nxt = B1;
    for (int t = 0; t < 5; ++t) {
        gemm_bf16<false><<<gS, 256, 0, stream>>>(cur, cur, nullptr, T1, nullptr,
                                                 Nn, Nn, Nn, 1.0f, 0.0f, 0.0f);
        gemm_bf16<false><<<gS, 256, 0, stream>>>(T1, cur, nullptr, T2, nullptr,
                                                 Nn, Nn, Nn, 1.0f, 0.0f, 0.0f);
        gemm_bf16<false><<<gS, 256, 0, stream>>>(T2, Sb, cur, nxt, nullptr,
                                                 Nn, Nn, Nn, -0.5f, 1.5f, 0.0f);
        unsigned short* tmp = cur; cur = nxt; nxt = tmp;
    }

    // 6) W = B V  -> f32 output
    dim3 gW(Kf / 128, Nn / 128);
    gemm_bf16<false><<<gW, 256, 0, stream>>>(cur, Vb, nullptr, nullptr, Wout,
                                             Nn, Kf, Nn, 1.0f, 0.0f, 0.0f);
}
